// Attention_11166914969728
// MI455X (gfx1250) — matrix operand order repaired and checked by reading, not compiled, not run on hardware
//
#include <hip/hip_runtime.h>
#include <hip/hip_bf16.h>
#include <math.h>

typedef __bf16 bf16;
typedef __attribute__((ext_vector_type(16))) __bf16 bf16x16;
typedef __attribute__((ext_vector_type(8)))  __bf16 bf16x8;
typedef __attribute__((ext_vector_type(8)))  float  f32x8;

#define WMMA_BF16(A, B, C) \
  __builtin_amdgcn_wmma_f32_16x16x32_bf16(false, (A), false, (B), (short)0, (C), false, false)

// ---------------------------------------------------------------------------
// Fragment loaders (bf16, 16x16x32 WMMA)
// A 16x32: lane l holds row M=l%16; elem e -> K=(e/8)*16+(l/16)*8+(e%8)
//   => two 16B chunks at k0+(l/16)*8 and k0+16+(l/16)*8 within the row.
// B 32x16: lane l holds col N=l%16; elem e -> K=(e/8)*16+(l/16)*8+(e%8)
//   => two 16B chunks at k0+(l/16)*8 and k0+16+(l/16)*8 within the "row" (col of B).
// C/D f32 16x16: vgpr r, lane l -> M=r+8*(l/16), N=l%16
// ---------------------------------------------------------------------------
__device__ __forceinline__ bf16x16 fragA_ld(const bf16* rowp, int half) {
  bf16x8 lo = *(const bf16x8*)(rowp + half * 8);
  bf16x8 hi = *(const bf16x8*)(rowp + 16 + half * 8);
  bf16x16 r;
#pragma unroll
  for (int i = 0; i < 8; ++i) { r[i] = lo[i]; r[8 + i] = hi[i]; }
  return r;
}

__device__ __forceinline__ bf16x16 fragB_ld(const bf16* rowp, int half) {
  bf16x8 lo = *(const bf16x8*)(rowp + half * 8);
  bf16x8 hi = *(const bf16x8*)(rowp + 16 + half * 8);
  bf16x16 r;
#pragma unroll
  for (int i = 0; i < 8; ++i) { r[i] = lo[i]; r[8 + i] = hi[i]; }
  return r;
}

// ---------------------------------------------------------------------------
// Kernel 1: fp32 -> bf16 conversion
// ---------------------------------------------------------------------------
__global__ void cvt_f32_to_bf16(const float* __restrict__ in, bf16* __restrict__ out, int n) {
  int i = blockIdx.x * blockDim.x + threadIdx.x;
  if (i < n) out[i] = (bf16)in[i];
}

// ---------------------------------------------------------------------------
// Kernel 2: gather relative-position bias  biasf[h][i][j] = table[idx[i,j]*12 + h]
// ---------------------------------------------------------------------------
__global__ void bias_gather(const float* __restrict__ table, const int* __restrict__ idx,
                            float* __restrict__ biasf) {
  int t = blockIdx.x * blockDim.x + threadIdx.x;
  if (t < 12 * 512 * 512) {
    int h  = t >> 18;
    int ij = t & 262143;
    biasf[t] = table[idx[ij] * 12 + h];
  }
}

// ---------------------------------------------------------------------------
// Kernel 3: QKV GEMM  out[m,c] = sum_k x[m,k] * Wqkv[c,k] + bqkv[c]
// M=16384, N=2304, K=768. Workgroup tile 128x128 (8 waves, each 32x64).
// Epilogue scatters q (scaled) -> [b,h,n,d], k -> [b,h,n,d], v -> [b,h,d,n].
// ---------------------------------------------------------------------------
__global__ void qkv_gemm(const bf16* __restrict__ xb, const bf16* __restrict__ wb,
                         const float* __restrict__ bias, bf16* __restrict__ qo,
                         bf16* __restrict__ ko, bf16* __restrict__ vTo) {
  const int lane = threadIdx.x & 31, wid = threadIdx.x >> 5;
  const int half = lane >> 4, l16 = lane & 15;
  const int mbase = blockIdx.x * 128 + (wid & 3) * 32;
  const int nbase = blockIdx.y * 128 + (wid >> 2) * 64;

  f32x8 acc[2][4] = {};
  for (int k0 = 0; k0 < 768; k0 += 32) {
    bf16x16 af[2], bfr[4];
#pragma unroll
    for (int mt = 0; mt < 2; ++mt)
      af[mt] = fragA_ld(xb + (size_t)(mbase + mt * 16 + l16) * 768 + k0, half);
#pragma unroll
    for (int nt = 0; nt < 4; ++nt)
      bfr[nt] = fragB_ld(wb + (size_t)(nbase + nt * 16 + l16) * 768 + k0, half);
#pragma unroll
    for (int mt = 0; mt < 2; ++mt)
#pragma unroll
      for (int nt = 0; nt < 4; ++nt)
        acc[mt][nt] = WMMA_BF16(af[mt], bfr[nt], acc[mt][nt]);
  }

#pragma unroll
  for (int mt = 0; mt < 2; ++mt)
#pragma unroll
    for (int nt = 0; nt < 4; ++nt) {
      const int c     = nbase + nt * 16 + l16;
      const int which = c / 768;
      const int rem   = c - which * 768;
      const int h = rem >> 6, d = rem & 63;
      const float bv = bias[c];
#pragma unroll
      for (int r = 0; r < 8; ++r) {
        const int m  = mbase + mt * 16 + r + 8 * half;
        const int bi = m >> 9, n = m & 511;
        const float v = acc[mt][nt][r] + bv;
        const size_t bh = (size_t)bi * 12 + h;
        if (which == 0)      qo[(bh * 512 + n) * 64 + d] = (bf16)(v * 0.125f);  // *SCALE
        else if (which == 1) ko[(bh * 512 + n) * 64 + d] = (bf16)v;
        else                 vTo[(bh * 64 + d) * 512 + n] = (bf16)v;            // transposed
      }
    }
}

// ---------------------------------------------------------------------------
// Kernel 4: flash attention per (b,h). 8 waves x 4 passes x 16 q-rows.
// Streams 32-wide key blocks: QK^T WMMA -> +bias -> online softmax ->
// P transposed through a PER-WAVE LDS slab -> PV WMMA against v^T.
// No block barriers: pbuf[wid] is wave-private and LDS ops from one wave are
// in-order (DScnt); a single s_wait_dscnt + compiler fence orders the
// transpose store->load. Waves drift independently through the j-loop,
// improving latency hiding of K/V/bias loads.
// ---------------------------------------------------------------------------
__global__ void attn_kernel(const bf16* __restrict__ qg, const bf16* __restrict__ kg,
                            const bf16* __restrict__ vTg, const float* __restrict__ biasf,
                            bf16* __restrict__ attnb) {
  __shared__ bf16 pbuf[8][16][32];  // per-wave P staging (transpose C-layout -> A-layout)

  const int bh = blockIdx.x;
  const int b = bh / 12, h = bh % 12;
  const bf16*  qp = qg  + (size_t)bh * 512 * 64;
  const bf16*  kp = kg  + (size_t)bh * 512 * 64;
  const bf16*  vp = vTg + (size_t)bh * 64 * 512;
  const float* bp = biasf + (size_t)h * 512 * 512;

  const int lane = threadIdx.x & 31, wid = threadIdx.x >> 5;
  const int half = lane >> 4, l16 = lane & 15;
  const float LOG2E = 1.44269504088896340736f;

  for (int pass = 0; pass < 4; ++pass) {
    const int m0 = (pass * 8 + wid) * 16;

    bf16x16 qa[2];
#pragma unroll
    for (int ks = 0; ks < 2; ++ks)
      qa[ks] = fragA_ld(qp + (size_t)(m0 + l16) * 64 + ks * 32, half);

    f32x8 O[4] = {};
    float mstat[8], lstat[8];
#pragma unroll
    for (int r = 0; r < 8; ++r) { mstat[r] = -INFINITY; lstat[r] = 0.0f; }

    for (int j = 0; j < 16; ++j) {
      const int jb = j * 32;

      // S = q @ k^T for a 16x32 block
      f32x8 s[2] = {};
#pragma unroll
      for (int nt = 0; nt < 2; ++nt)
#pragma unroll
        for (int ks = 0; ks < 2; ++ks) {
          bf16x16 kf = fragB_ld(kp + (size_t)(jb + nt * 16 + l16) * 64 + ks * 32, half);
          s[nt] = WMMA_BF16(qa[ks], kf, s[nt]);
        }

      // + relative position bias
#pragma unroll
      for (int nt = 0; nt < 2; ++nt)
#pragma unroll
        for (int r = 0; r < 8; ++r)
          s[nt][r] += bp[(size_t)(m0 + r + 8 * half) * 512 + jb + nt * 16 + l16];

      // online softmax over this 32-column block (rows span 16-lane halves)
      float corr[8];
#pragma unroll
      for (int r = 0; r < 8; ++r) {
        float bm = fmaxf(s[0][r], s[1][r]);
        bm = fmaxf(bm, __shfl_xor(bm, 1, 32));
        bm = fmaxf(bm, __shfl_xor(bm, 2, 32));
        bm = fmaxf(bm, __shfl_xor(bm, 4, 32));
        bm = fmaxf(bm, __shfl_xor(bm, 8, 32));
        const float mnew = fmaxf(mstat[r], bm);
        corr[r] = exp2f((mstat[r] - mnew) * LOG2E);
        mstat[r] = mnew;
        const float p0 = exp2f((s[0][r] - mnew) * LOG2E);
        const float p1 = exp2f((s[1][r] - mnew) * LOG2E);
        s[0][r] = p0; s[1][r] = p1;
        float bs = p0 + p1;
        bs += __shfl_xor(bs, 1, 32);
        bs += __shfl_xor(bs, 2, 32);
        bs += __shfl_xor(bs, 4, 32);
        bs += __shfl_xor(bs, 8, 32);
        lstat[r] = lstat[r] * corr[r] + bs;
      }

      // rescale O
#pragma unroll
      for (int nd = 0; nd < 4; ++nd)
#pragma unroll
        for (int r = 0; r < 8; ++r) O[nd][r] *= corr[r];

      // P (C-layout) -> wave-private LDS slab, row-major
#pragma unroll
      for (int nt = 0; nt < 2; ++nt)
#pragma unroll
        for (int r = 0; r < 8; ++r)
          pbuf[wid][r + 8 * half][nt * 16 + l16] = (bf16)s[nt][r];

      // Order the wave's own LDS stores before the transposed reads.
      // (pbuf[wid] is private to this wave; LDS is in-order per wave, so a
      // DScnt wait + compiler fence replaces block-wide __syncthreads.)
      asm volatile("s_wait_dscnt 0x0" ::: "memory");

      // P as A fragment, v^T rows as B fragments; O += P @ V
      bf16x16 pa = fragA_ld(&pbuf[wid][l16][0], half);
#pragma unroll
      for (int nd = 0; nd < 4; ++nd) {
        bf16x16 vf = fragB_ld(vp + (size_t)(nd * 16 + l16) * 512 + jb, half);
        O[nd] = WMMA_BF16(pa, vf, O[nd]);
      }
      // No trailing barrier needed: next iteration's stores are later LDS ops
      // from this same wave (in-order), and the pa VGPRs are already consumed.
    }

    // normalize and store to [b, n, h*64 + d] (bf16)
#pragma unroll
    for (int nd = 0; nd < 4; ++nd)
#pragma unroll
      for (int r = 0; r < 8; ++r) {
        const int tok = m0 + r + 8 * half;
        const int d   = nd * 16 + l16;
        attnb[((size_t)b * 512 + tok) * 768 + h * 64 + d] = (bf16)(O[nd][r] / lstat[r]);
      }
  }
}

// ---------------------------------------------------------------------------
// Kernel 5: output projection  out[m,c] = sum_k y[m,k] * Wproj[c,k] + bproj[c]
// M=16384, N=768, K=768 -> fp32 d_out
// ---------------------------------------------------------------------------
__global__ void proj_gemm(const bf16* __restrict__ yb, const bf16* __restrict__ wb,
                          const float* __restrict__ bias, float* __restrict__ out) {
  const int lane = threadIdx.x & 31, wid = threadIdx.x >> 5;
  const int half = lane >> 4, l16 = lane & 15;
  const int mbase = blockIdx.x * 128 + (wid & 3) * 32;
  const int nbase = blockIdx.y * 128 + (wid >> 2) * 64;

  f32x8 acc[2][4] = {};
  for (int k0 = 0; k0 < 768; k0 += 32) {
    bf16x16 af[2], bfr[4];
#pragma unroll
    for (int mt = 0; mt < 2; ++mt)
      af[mt] = fragA_ld(yb + (size_t)(mbase + mt * 16 + l16) * 768 + k0, half);
#pragma unroll
    for (int nt = 0; nt < 4; ++nt)
      bfr[nt] = fragB_ld(wb + (size_t)(nbase + nt * 16 + l16) * 768 + k0, half);
#pragma unroll
    for (int mt = 0; mt < 2; ++mt)
#pragma unroll
      for (int nt = 0; nt < 4; ++nt)
        acc[mt][nt] = WMMA_BF16(af[mt], bfr[nt], acc[mt][nt]);
  }

#pragma unroll
  for (int mt = 0; mt < 2; ++mt)
#pragma unroll
    for (int nt = 0; nt < 4; ++nt) {
      const int c = nbase + nt * 16 + l16;
      const float bv = bias[c];
#pragma unroll
      for (int r = 0; r < 8; ++r) {
        const int m = mbase + mt * 16 + r + 8 * half;
        out[(size_t)m * 768 + c] = acc[mt][nt][r] + bv;
      }
    }
}

// ---------------------------------------------------------------------------
// Launcher
// ---------------------------------------------------------------------------
extern "C" void kernel_launch(void* const* d_in, const int* in_sizes, int n_in,
                              void* d_out, int out_size, void* d_ws, size_t ws_size,
                              hipStream_t stream) {
  const float* x     = (const float*)d_in[0];  // [32, 512, 768]
  const float* Wqkv  = (const float*)d_in[1];  // [2304, 768]
  const float* bqkv  = (const float*)d_in[2];  // [2304]
  const float* table = (const float*)d_in[3];  // [3375, 12]
  const int*   rpidx = (const int*)d_in[4];    // [512, 512]
  const float* Wproj = (const float*)d_in[5];  // [768, 768]
  const float* bproj = (const float*)d_in[6];  // [768]
  float* out = (float*)d_out;

  // workspace layout (bytes)
  constexpr size_t SZ_XB    = (size_t)16384 * 768 * 2;      // x bf16
  constexpr size_t SZ_WQKV  = (size_t)2304 * 768 * 2;       // Wqkv bf16
  constexpr size_t SZ_WPROJ = (size_t)768 * 768 * 2;        // Wproj bf16
  constexpr size_t SZ_QKV1  = (size_t)384 * 512 * 64 * 2;   // one of q/k/vT bf16
  constexpr size_t SZ_ATTN  = (size_t)16384 * 768 * 2;      // attention output bf16
  constexpr size_t SZ_BIAS  = (size_t)12 * 512 * 512 * 4;   // gathered bias fp32
  constexpr size_t NEEDED = SZ_XB + SZ_WQKV + SZ_WPROJ + 3 * SZ_QKV1 + SZ_ATTN + SZ_BIAS;
  if (ws_size < NEEDED) return;

  char* ws = (char*)d_ws;
  bf16*  xb    = (bf16*)(ws);
  bf16*  wqkvb = (bf16*)(ws + SZ_XB);
  bf16*  wprjb = (bf16*)(ws + SZ_XB + SZ_WQKV);
  bf16*  qb    = (bf16*)(ws + SZ_XB + SZ_WQKV + SZ_WPROJ);
  bf16*  kb    = (bf16*)(ws + SZ_XB + SZ_WQKV + SZ_WPROJ + SZ_QKV1);
  bf16*  vTb   = (bf16*)(ws + SZ_XB + SZ_WQKV + SZ_WPROJ + 2 * SZ_QKV1);
  bf16*  attnb = (bf16*)(ws + SZ_XB + SZ_WQKV + SZ_WPROJ + 3 * SZ_QKV1);
  float* biasf = (float*)(ws + SZ_XB + SZ_WQKV + SZ_WPROJ + 3 * SZ_QKV1 + SZ_ATTN);

  // 1) convert inputs to bf16
  {
    int n = 16384 * 768;
    cvt_f32_to_bf16<<<(n + 255) / 256, 256, 0, stream>>>(x, xb, n);
    n = 2304 * 768;
    cvt_f32_to_bf16<<<(n + 255) / 256, 256, 0, stream>>>(Wqkv, wqkvb, n);
    n = 768 * 768;
    cvt_f32_to_bf16<<<(n + 255) / 256, 256, 0, stream>>>(Wproj, wprjb, n);
  }

  // 2) gather relative-position bias into [12, 512, 512] fp32
  {
    int n = 12 * 512 * 512;
    bias_gather<<<(n + 255) / 256, 256, 0, stream>>>(table, rpidx, biasf);
  }

  // 3) QKV projection (WMMA)
  qkv_gemm<<<dim3(16384 / 128, 2304 / 128), 256, 0, stream>>>(xb, wqkvb, bqkv, qb, kb, vTb);

  // 4) attention (WMMA flash)
  attn_kernel<<<384, 256, 0, stream>>>(qb, kb, vTb, biasf, attnb);

  // 5) output projection (WMMA) -> fp32 out
  proj_gemm<<<dim3(16384 / 128, 768 / 128), 256, 0, stream>>>(attnb, wprjb, bproj, out);
}
